// GraphUnetNoPool_84808424227301
// MI455X (gfx1250) — matrix operand order repaired and checked by reading, not compiled, not run on hardware
//
#include <hip/hip_runtime.h>
#include <hip/hip_bf16.h>

typedef __bf16 bf16;
typedef __attribute__((ext_vector_type(16))) __bf16 v16bf;
typedef __attribute__((ext_vector_type(8)))  __bf16 v8bf;
typedef __attribute__((ext_vector_type(4)))  __bf16 v4bf;
typedef __attribute__((ext_vector_type(8)))  float  v8f;

static constexpr int NN = 8192;   // nodes
static constexpr int DD = 256;    // feature dim

// ---------------------------------------------------------------------------
// deg[i] = 1 + sum_j g[i][j];  dinv[i] = deg>0 ? rsqrt(deg) : 0
// ---------------------------------------------------------------------------
__global__ __launch_bounds__(256) void k_rowsum_rsqrt(const float* __restrict__ g,
                                                      float* __restrict__ dinv) {
  const int row = blockIdx.x;
  const float* gr = g + (size_t)row * NN;
  float s = 0.f;
  for (int j = threadIdx.x * 4; j < NN; j += 256 * 4) {
    float4 v = *(const float4*)(gr + j);
    s += v.x + v.y + v.z + v.w;
  }
  __shared__ float red[256];
  red[threadIdx.x] = s;
  __syncthreads();
  for (int off = 128; off > 0; off >>= 1) {
    if (threadIdx.x < off) red[threadIdx.x] += red[threadIdx.x + off];
    __syncthreads();
  }
  if (threadIdx.x == 0) {
    float d = red[0] + 1.0f;
    dinv[row] = d > 0.f ? rsqrtf(d) : 0.f;
  }
}

// ---------------------------------------------------------------------------
// gn[i][j] = bf16( (g[i][j] + (i==j)) * dinv[i] * dinv[j] )   (4 elems/thread)
// ---------------------------------------------------------------------------
__global__ __launch_bounds__(256) void k_normalize(const float* __restrict__ g,
                                                   const float* __restrict__ dinv,
                                                   bf16* __restrict__ gn) {
  const size_t base = ((size_t)blockIdx.x * 256 + threadIdx.x) * 4;
  const int i = (int)(base >> 13);       // row (NN = 8192 = 2^13)
  const int j = (int)(base & (NN - 1));  // col
  const float di = dinv[i];
  float4 v = *(const float4*)(g + base);
  v4bf o;
  o[0] = (bf16)((v.x + (float)(j + 0 == i)) * di * dinv[j + 0]);
  o[1] = (bf16)((v.y + (float)(j + 1 == i)) * di * dinv[j + 1]);
  o[2] = (bf16)((v.z + (float)(j + 2 == i)) * di * dinv[j + 2]);
  o[3] = (bf16)((v.w + (float)(j + 3 == i)) * di * dinv[j + 3]);
  *(v4bf*)(gn + base) = o;
}

// ---------------------------------------------------------------------------
// elementwise helpers
// ---------------------------------------------------------------------------
__global__ void k_f2b(const float* __restrict__ in, bf16* __restrict__ out, int n) {
  int i = blockIdx.x * 256 + threadIdx.x;
  if (i < n) out[i] = (bf16)in[i];
}
__global__ void k_add_f2b(const float* __restrict__ a, const float* __restrict__ b,
                          bf16* __restrict__ out, int n) {
  int i = blockIdx.x * 256 + threadIdx.x;
  if (i < n) out[i] = (bf16)(a[i] + b[i]);
}
__global__ void k_add_f(const float* __restrict__ a, const float* __restrict__ b,
                        float* __restrict__ out, int n) {
  int i = blockIdx.x * 256 + threadIdx.x;
  if (i < n) out[i] = a[i] + b[i];
}

// ---------------------------------------------------------------------------
// Tiled bf16 WMMA GEMM:  C[M,256] = A[M,K] * B  (+bias, relu)
//   BT=false : B is [K,256] row-major        (C = A @ B)     — aggregation
//   BT=true  : B is [256,K] row-major (W)    (C = A @ B^T)   — linear layer
// Block tile 128x64, BK=64, double-buffered LDS (48 KB of the 320 KB WGP LDS).
// 8 waves (wave32); each wave owns a 32x32 tile = 2x2 accumulators;
// 8 v_wmma_f32_16x16x32_bf16 per chunk between barriers, with the next
// chunk's global loads in flight behind the WMMAs.
// ---------------------------------------------------------------------------
template <bool BT>
__global__ __launch_bounds__(256) void k_gemm(const bf16* __restrict__ A, int lda,
                                              const bf16* __restrict__ B, int ldb,
                                              const float* __restrict__ bias,
                                              float* __restrict__ outF,
                                              bf16* __restrict__ outB,
                                              int K, int relu) {
  __shared__ __align__(16) bf16 As[2][128 * 64];  // [m][k] row-major
  __shared__ __align__(16) bf16 Bt[2][64 * 64];   // [n][k] row-major (B transposed)

  const int tid  = threadIdx.x;
  const int lane = tid & 31;
  const int wv   = tid >> 5;
  const int wm   = wv & 3;        // wave row (4)
  const int wn   = wv >> 2;       // wave col (2)

  const int m0 = blockIdx.x * 128;
  const int n0 = blockIdx.y * 64;

  // A loader: rows ar, ar+64; 16-elem chunks along k
  const int ar = tid >> 2;            // 0..63
  const int ac = (tid & 3) * 16;      // 0,16,32,48
  // B loader: rows bk, bk+32 (k rows if !BT, W/n rows if BT); 8-elem chunks
  const int bk = tid >> 3;            // 0..31
  const int bc = (tid & 7) * 8;       // 0..56

  const bf16* aP0 = A + (size_t)(m0 + ar) * lda + ac;
  const bf16* aP1 = A + (size_t)(m0 + ar + 64) * lda + ac;
  const bf16* bP0;
  const bf16* bP1;
  if (BT) {
    bP0 = B + (size_t)(n0 + bk) * ldb + bc;
    bP1 = B + (size_t)(n0 + bk + 32) * ldb + bc;
  } else {
    bP0 = B + (size_t)bk * ldb + n0 + bc;
    bP1 = B + (size_t)(bk + 32) * ldb + n0 + bc;
  }
  const size_t aStep = 64;
  const size_t bStep = BT ? (size_t)64 : (size_t)64 * ldb;

  v8f acc[2][2] = {};
  v8bf ra0, ra1, ra2, ra3, rb0, rb1;

  auto gload = [&]() {
    ra0 = *(const v8bf*)(aP0);
    ra1 = *(const v8bf*)(aP0 + 8);
    ra2 = *(const v8bf*)(aP1);
    ra3 = *(const v8bf*)(aP1 + 8);
    rb0 = *(const v8bf*)(bP0);
    rb1 = *(const v8bf*)(bP1);
    aP0 += aStep; aP1 += aStep; bP0 += bStep; bP1 += bStep;
  };

  auto sstore = [&](int buf) {
    *(v8bf*)&As[buf][ar * 64 + ac]            = ra0;
    *(v8bf*)&As[buf][ar * 64 + ac + 8]        = ra1;
    *(v8bf*)&As[buf][(ar + 64) * 64 + ac]     = ra2;
    *(v8bf*)&As[buf][(ar + 64) * 64 + ac + 8] = ra3;
    if (BT) {
      *(v8bf*)&Bt[buf][bk * 64 + bc]        = rb0;   // W rows already [n][k]
      *(v8bf*)&Bt[buf][(bk + 32) * 64 + bc] = rb1;
    } else {
      #pragma unroll
      for (int e = 0; e < 8; ++e) {                  // transpose into [n][k]
        Bt[buf][(bc + e) * 64 + bk]      = rb0[e];
        Bt[buf][(bc + e) * 64 + bk + 32] = rb1[e];
      }
    }
  };

  // per-lane fragment indices (CDNA5 wave32 WMMA layouts)
  const int lrow = lane & 15;
  const int ka = (lane >> 4) * 8;   // A: lanes 0-15 K{0..7,16..23}, 16-31 K{8..15,24..31}
  const int kb = (lane >> 4) * 8;   // B: lanes 0-15 K{0..7,16..23}, 16-31 K{8..15,24..31}

  auto compute = [&](int buf) {
    #pragma unroll
    for (int s = 0; s < 2; ++s) {
      const int koff = s * 32;
      union Frag { v16bf v; v8bf h[2]; };
      Frag af[2], bfg[2];
      #pragma unroll
      for (int mf = 0; mf < 2; ++mf) {
        const int r = wm * 32 + mf * 16 + lrow;
        af[mf].h[0] = *(const v8bf*)&As[buf][r * 64 + koff + ka];
        af[mf].h[1] = *(const v8bf*)&As[buf][r * 64 + koff + 16 + ka];
      }
      #pragma unroll
      for (int nf = 0; nf < 2; ++nf) {
        const int c = wn * 32 + nf * 16 + lrow;
        bfg[nf].h[0] = *(const v8bf*)&Bt[buf][c * 64 + koff + kb];
        bfg[nf].h[1] = *(const v8bf*)&Bt[buf][c * 64 + koff + 16 + kb];
      }
      #pragma unroll
      for (int mf = 0; mf < 2; ++mf)
        #pragma unroll
        for (int nf = 0; nf < 2; ++nf)
          acc[mf][nf] = __builtin_amdgcn_wmma_f32_16x16x32_bf16(
              false, af[mf].v, false, bfg[nf].v, (short)0, acc[mf][nf], false, false);
    }
  };

  const int nch = K >> 6;          // chunks of 64
  gload();                          // chunk 0 -> regs
  sstore(0);                        // regs -> buf 0
  for (int ch = 0; ch < nch; ++ch) {
    const int cur = ch & 1;
    const bool more = (ch + 1) < nch;
    if (more) {
      __builtin_prefetch(aP0 + 64, 0, 0);   // chunk ch+2 toward caches
      gload();                              // chunk ch+1 -> regs (in flight)
    }
    __syncthreads();                        // publish buf[cur]
    compute(cur);                           // 8 WMMAs, loads hidden behind them
    if (more) sstore(cur ^ 1);              // regs -> other buffer
  }

  // epilogue: C layout — lanes 0-15: N=lane, M=v; lanes 16-31: N=lane-16, M=v+8
  const int half8 = (lane >> 4) * 8;
  #pragma unroll
  for (int mf = 0; mf < 2; ++mf) {
    #pragma unroll
    for (int nf = 0; nf < 2; ++nf) {
      const int col = n0 + wn * 32 + nf * 16 + lrow;
      const float bb = bias ? bias[col] : 0.f;
      #pragma unroll
      for (int v = 0; v < 8; ++v) {
        const int row = m0 + wm * 32 + mf * 16 + half8 + v;
        float x = acc[mf][nf][v] + bb;
        if (relu) x = x > 0.f ? x : 0.f;
        const size_t o = (size_t)row * DD + col;
        if (outF) outF[o] = x;
        if (outB) outB[o] = (bf16)x;
      }
    }
  }
}

// ---------------------------------------------------------------------------
extern "C" void kernel_launch(void* const* d_in, const int* in_sizes, int n_in,
                              void* d_out, int out_size, void* d_ws, size_t ws_size,
                              hipStream_t stream) {
  const float* g    = (const float*)d_in[0];
  const float* h    = (const float*)d_in[1];
  const float* Wd   = (const float*)d_in[2];  // [3,256,256]
  const float* bd   = (const float*)d_in[3];  // [3,256]
  const float* Wbm  = (const float*)d_in[4];  // [256,256]
  const float* bbm  = (const float*)d_in[5];  // [256]
  const float* Wu   = (const float*)d_in[6];  // [3,256,256]
  const float* bu   = (const float*)d_in[7];  // [3,256]
  float* out = (float*)d_out;                 // [4, 8192, 256]

  // workspace carve-out
  char* p = (char*)d_ws;
  auto take = [&](size_t bytes) -> char* {
    char* q = p;
    p += (bytes + 255) & ~(size_t)255;
    return q;
  };
  bf16*  GN   = (bf16*)take((size_t)NN * NN * 2);   // 128 MB — fits 192 MB L2
  bf16*  curb = (bf16*)take((size_t)NN * DD * 2);   // current h (bf16)
  bf16*  aggb = (bf16*)take((size_t)NN * DD * 2);   // GN @ h (bf16)
  float* dn1  = (float*)take((size_t)NN * DD * 4);
  float* dn2  = (float*)take((size_t)NN * DD * 4);
  float* dn3  = (float*)take((size_t)NN * DD * 4);
  float* hb   = (float*)take((size_t)NN * DD * 4);
  float* dinv = (float*)take((size_t)NN * 4);
  bf16*  Wb   = (bf16*)take((size_t)7 * DD * DD * 2);
  bf16*  WbD  = Wb;               // 3 matrices
  bf16*  WbB  = Wb + 3 * DD * DD; // 1 matrix
  bf16*  WbU  = Wb + 4 * DD * DD; // 3 matrices

  const int nd = NN * DD;

  // 1) normalization + precision conversion (one-time, bandwidth-bound)
  k_rowsum_rsqrt<<<NN, 256, 0, stream>>>(g, dinv);
  k_normalize<<<(NN / 4) * (NN / 256), 256, 0, stream>>>(g, dinv, GN);
  k_f2b<<<nd / 256, 256, 0, stream>>>(h, curb, nd);
  k_f2b<<<(3 * DD * DD) / 256, 256, 0, stream>>>(Wd, WbD, 3 * DD * DD);
  k_f2b<<<(DD * DD) / 256, 256, 0, stream>>>(Wbm, WbB, DD * DD);
  k_f2b<<<(3 * DD * DD) / 256, 256, 0, stream>>>(Wu, WbU, 3 * DD * DD);

  // 2) GCN layers (all matmuls on the bf16 WMMA path, fp32 accumulate)
  dim3 grid(NN / 128, DD / 64);
  auto gcn = [&](const bf16* Wl, const float* bl, float* of, bf16* ob) {
    // agg = GN @ h  (K = 8192, A streams from L2 after first pass)
    k_gemm<false><<<grid, 256, 0, stream>>>(GN, NN, curb, DD, nullptr,
                                            nullptr, aggb, NN, 0);
    // out = relu(agg @ W^T + b)  (K = 256)
    k_gemm<true><<<grid, 256, 0, stream>>>(aggb, DD, Wl, DD, bl,
                                           of, ob, DD, 1);
  };

  // down
  gcn(WbD + 0 * DD * DD, bd + 0 * DD, dn1, curb);
  gcn(WbD + 1 * DD * DD, bd + 1 * DD, dn2, curb);
  gcn(WbD + 2 * DD * DD, bd + 2 * DD, dn3, curb);
  // bottom
  gcn(WbB, bbm, hb, nullptr);
  // up with skip connections
  k_add_f2b<<<nd / 256, 256, 0, stream>>>(hb, dn3, curb, nd);
  gcn(WbU + 0 * DD * DD, bu + 0 * DD, out + 0 * (size_t)nd, nullptr);
  k_add_f2b<<<nd / 256, 256, 0, stream>>>(out + 0 * (size_t)nd, dn2, curb, nd);
  gcn(WbU + 1 * DD * DD, bu + 1 * DD, out + 1 * (size_t)nd, nullptr);
  k_add_f2b<<<nd / 256, 256, 0, stream>>>(out + 1 * (size_t)nd, dn1, curb, nd);
  gcn(WbU + 2 * DD * DD, bu + 2 * DD, out + 2 * (size_t)nd, nullptr);
  // hs[3] = hs[2] + org_h
  k_add_f<<<nd / 256, 256, 0, stream>>>(out + 2 * (size_t)nd, h, out + 3 * (size_t)nd, nd);
}
